// LCAOConv_14955076125266
// MI455X (gfx1250) — matrix operand order repaired and checked by reading, not compiled, not run on hardware
//
#include <hip/hip_runtime.h>
#include <math.h>

// LCAO graph-conv for gfx1250 (MI455X), wave32 + WMMA, vectorized fragments.
// Pipeline:
//  P0 pack_b     : pre-pack all 7 weight matrices into f16 B-fragment layout
//  K0 zero       : agg[E,64], node_acc[N,64]
//  K1 node_before: xq,xk = x@W_nb+b              (WMMA)
//  K2 coeffs     : c = silu(silu(cji)@Wc1)@Wc2 -> cji2 (f16), l2norm(ckj) (f16)
//  K3 triplets   : gather ckj_norm, weighted D-dot, l2norm, *sigmoid(xk[k]), scatter->agg
//  K4 three_lin  : tbw2 = silu(agg)@W_t3+b       (WMMA)
//  K5 lcao       : per-(e,d) l2norm(cji2*(1+tbw2)), robs-dot, l2norm -> lcao
//  K6 node_msg   : m = MLP(silu(concat gather xq)) (WMMA); scatter lcao*m -> node_acc
//  K7 out        : out = x + node_acc@W_na       (WMMA)
//
// Note on row guards: A-fragment rows are only address-clamped (never value-
// masked) — in a GEMM an A row affects only its own output row, and all
// out-of-range output rows are store-guarded. This avoids 16 v_cndmask per
// fragment.

typedef __attribute__((ext_vector_type(16))) _Float16 v16h;
typedef __attribute__((ext_vector_type(8)))  _Float16 v8h;
typedef __attribute__((ext_vector_type(2)))  _Float16 v2h;
typedef __attribute__((ext_vector_type(8)))  float    v8f;
typedef __attribute__((ext_vector_type(4)))  float    v4f;
typedef __attribute__((ext_vector_type(2)))  float    v2f;

#define DEV __device__ __forceinline__

DEV float sigmoidf_(float x) { return 1.0f / (1.0f + __expf(-x)); }
DEV float siluf_(float x)    { return x * sigmoidf_(x); }

DEV v8f vzero8() {
  v8f v;
#pragma unroll
  for (int i = 0; i < 8; ++i) v[i] = 0.0f;
  return v;
}

DEV v8f wmma16(v16h a, v16h b, v8f c) {
  return __builtin_amdgcn_wmma_f32_16x16x32_f16(false, a, false, b, (short)0, c,
                                                false, false);
}

// ---- A fragment: 16x32 f16 (ISA 7.12.2). Per lane the 16 K-values are two
// contiguous 8-wide runs: [k0+hi*8, +7] and [k0+16+hi*8, +7]. 4x float4 loads.
template <bool SILU>
DEV v16h load_a_f32(const float* __restrict__ A, long row0, long nrows, int lda,
                    int k0) {
  int lane = threadIdx.x & 31;
  int m = lane & 15, hi = lane >> 4;
  long r = row0 + m;
  if (r >= nrows) r = row0;  // address clamp only; stores are row-guarded
  const float* rp = A + r * (long)lda + k0 + hi * 8;
  v4f q0 = *(const v4f*)(rp);
  v4f q1 = *(const v4f*)(rp + 4);
  v4f q2 = *(const v4f*)(rp + 16);
  v4f q3 = *(const v4f*)(rp + 20);
  v16h a;
#pragma unroll
  for (int i = 0; i < 4; ++i) {
    float f0 = q0[i], f1 = q1[i], f2 = q2[i], f3 = q3[i];
    if (SILU) { f0 = siluf_(f0); f1 = siluf_(f1); f2 = siluf_(f2); f3 = siluf_(f3); }
    a[i] = (_Float16)f0;  a[4 + i]  = (_Float16)f1;
    a[8 + i] = (_Float16)f2; a[12 + i] = (_Float16)f3;
  }
  return a;
}

// A fragment for silu(concat(xq[idx_i[row]], xq[idx_j[row]])), k0 in {0,32,64,96}
DEV v16h load_a_concat_silu(const float* __restrict__ xq,
                            const int* __restrict__ ii,
                            const int* __restrict__ jj, long row0, long rows,
                            int k0) {
  int lane = threadIdx.x & 31;
  int m = lane & 15, hi = lane >> 4;
  long r = row0 + m;
  if (r >= rows) r = row0;  // address clamp only
  const int* idx = (k0 < 64) ? ii : jj;
  int cbase = ((k0 < 64) ? k0 : (k0 - 64)) + hi * 8;
  long node = (long)idx[r];
  const float* rp = xq + node * 64 + cbase;
  v4f q0 = *(const v4f*)(rp);
  v4f q1 = *(const v4f*)(rp + 4);
  v4f q2 = *(const v4f*)(rp + 16);
  v4f q3 = *(const v4f*)(rp + 20);
  v16h a;
#pragma unroll
  for (int i = 0; i < 4; ++i) {
    a[i]      = (_Float16)siluf_(q0[i]);
    a[4 + i]  = (_Float16)siluf_(q1[i]);
    a[8 + i]  = (_Float16)siluf_(q2[i]);
    a[12 + i] = (_Float16)siluf_(q3[i]);
  }
  return a;
}

// A fragment from LDS f16 strip: two contiguous 8-half runs -> 2x ds_load_b128
DEV v16h load_a_lds(const _Float16* hb, int stride, int k0) {
  int lane = threadIdx.x & 31;
  int m = lane & 15, hi = lane >> 4;
  const _Float16* rp = hb + m * stride + k0 + hi * 8;
  v8h c0 = *(const v8h*)(rp);
  v8h c1 = *(const v8h*)(rp + 16);
  v16h a;
#pragma unroll
  for (int i = 0; i < 8; ++i) { a[i] = c0[i]; a[8 + i] = c1[i]; }
  return a;
}

// ---- packed B fragments: Bp[((k0/32)*tiles + j)*32 + lane] is the lane's v16h
DEV v16h load_b_packed(const _Float16* __restrict__ Bp, int tiles, int k0,
                       int j) {
  int lane = threadIdx.x & 31;
  const v16h* p = (const v16h*)Bp;
  return p[((k0 >> 5) * tiles + j) * 32 + lane];
}

// ================= kernels =================

// pack B (KxNcols f32, row-major) into f16 B-fragment layout
__global__ void pack_b_kernel(const float* __restrict__ B,
                              _Float16* __restrict__ Bp, int K, int Ncols) {
  int wave = threadIdx.x >> 5, lane = threadIdx.x & 31;
  int tiles = Ncols >> 4, ksteps = K >> 5;
  int idx = blockIdx.x * (blockDim.x >> 5) + wave;
  if (idx >= ksteps * tiles) return;
  int ks = idx / tiles, j = idx % tiles;
  int n = lane & 15, hi = lane >> 4;
  _Float16* outp = Bp + ((long)idx * 32 + lane) * 16;
  const float* bp = B + (long)(ks * 32 + hi * 8) * Ncols + j * 16 + n;
#pragma unroll
  for (int q = 0; q < 16; ++q) outp[q] = (_Float16)bp[(long)(q < 8 ? q : q + 8) * Ncols];
}

__global__ void zero_kernel(float* __restrict__ p, int n) {
  int i = blockIdx.x * blockDim.x + threadIdx.x;
  if (i < n) p[i] = 0.0f;
}

// h = x @ W_nb + b_nb ; xq = h[:, :64], xk = h[:, 64:]
__global__ void node_before_kernel(const float* __restrict__ x,
                                   const _Float16* __restrict__ Wp,
                                   const float* __restrict__ b_nb,
                                   float* __restrict__ xq,
                                   float* __restrict__ xk, int N) {
  int wave = threadIdx.x >> 5, lane = threadIdx.x & 31;
  long strip = (long)blockIdx.x * (blockDim.x >> 5) + wave;
  long row0 = strip * 16;
  if (row0 >= N) return;
  v8f acc[8];
#pragma unroll
  for (int j = 0; j < 8; ++j) acc[j] = vzero8();
#pragma unroll
  for (int k0 = 0; k0 < 128; k0 += 32) {
    v16h a = load_a_f32<false>(x, row0, N, 128, k0);
#pragma unroll
    for (int j = 0; j < 8; ++j)
      acc[j] = wmma16(a, load_b_packed(Wp, 8, k0, j), acc[j]);
  }
  int n = lane & 15, hi = lane >> 4;
#pragma unroll
  for (int j = 0; j < 8; ++j) {
    int col = j * 16 + n;
    float bias = b_nb[col];
#pragma unroll
    for (int v = 0; v < 8; ++v) {
      long r = row0 + v + 8 * hi;
      if (r < N) {
        float val = acc[j][v] + bias;
        if (j < 4) xq[r * 64 + col] = val;
        else       xk[r * 64 + (col - 64)] = val;
      }
    }
  }
}

// rows = E*D. c = silu(silu(cji)@W_c1)@W_c2 ; cji2=c[:,:64], ckjn=l2norm(c[:,64:])
__global__ void coeffs_kernel(const float* __restrict__ cji,
                              const _Float16* __restrict__ Wc1p,
                              const _Float16* __restrict__ Wc2p,
                              _Float16* __restrict__ cji2,
                              _Float16* __restrict__ ckjn, long rows) {
  __shared__ alignas(16) _Float16 hbuf[4][16 * 72];
  int wave = threadIdx.x >> 5, lane = threadIdx.x & 31;
  long strip = (long)blockIdx.x * (blockDim.x >> 5) + wave;
  long row0 = strip * 16;
  if (row0 >= rows) return;
  _Float16* hb = hbuf[wave];
  int n = lane & 15, hi = lane >> 4;

  v8f acc1[4];
#pragma unroll
  for (int j = 0; j < 4; ++j) acc1[j] = vzero8();
#pragma unroll
  for (int k0 = 0; k0 < 64; k0 += 32) {
    v16h a = load_a_f32<true>(cji, row0, rows, 64, k0);  // silu(cji)
#pragma unroll
    for (int j = 0; j < 4; ++j)
      acc1[j] = wmma16(a, load_b_packed(Wc1p, 4, k0, j), acc1[j]);
  }
#pragma unroll
  for (int j = 0; j < 4; ++j)
#pragma unroll
    for (int v = 0; v < 8; ++v)
      hb[(v + 8 * hi) * 72 + j * 16 + n] = (_Float16)siluf_(acc1[j][v]);

  v8f acc2[8];
#pragma unroll
  for (int j = 0; j < 8; ++j) acc2[j] = vzero8();
#pragma unroll
  for (int k0 = 0; k0 < 64; k0 += 32) {
    v16h a = load_a_lds(hb, 72, k0);
#pragma unroll
    for (int j = 0; j < 8; ++j)
      acc2[j] = wmma16(a, load_b_packed(Wc2p, 8, k0, j), acc2[j]);
  }
#pragma unroll
  for (int j = 0; j < 4; ++j)
#pragma unroll
    for (int v = 0; v < 8; ++v) {
      long r = row0 + v + 8 * hi;
      if (r < rows) cji2[r * 64 + j * 16 + n] = (_Float16)acc2[j][v];
    }
#pragma unroll
  for (int v = 0; v < 8; ++v) {
    float ss = 0.0f;
#pragma unroll
    for (int j = 4; j < 8; ++j) ss += acc2[j][v] * acc2[j][v];
    ss += __shfl_xor(ss, 1, 32);
    ss += __shfl_xor(ss, 2, 32);
    ss += __shfl_xor(ss, 4, 32);
    ss += __shfl_xor(ss, 8, 32);
    float s = 1.0f / fmaxf(sqrtf(ss), 1e-12f);
    long r = row0 + v + 8 * hi;
    if (r < rows) {
#pragma unroll
      for (int j = 4; j < 8; ++j)
        ckjn[r * 64 + (j - 4) * 16 + n] = (_Float16)(acc2[j][v] * s);
    }
  }
}

// one wave per triplet; lane owns cols 2*lane, 2*lane+1
__global__ void triplet_kernel(const _Float16* __restrict__ ckjn,
                               const float* __restrict__ robs,
                               const float* __restrict__ shbs,
                               const int* __restrict__ tri_k,
                               const int* __restrict__ e_kj,
                               const int* __restrict__ e_ji,
                               const float* __restrict__ xk,
                               float* __restrict__ agg, int T) {
  int wave = threadIdx.x >> 5, lane = threadIdx.x & 31;
  long t = (long)blockIdx.x * (blockDim.x >> 5) + wave;
  if (t >= T) return;
  long e = e_kj[t];
  long kn = tri_k[t];
  long ej = e_ji[t];
  float a0 = 0.0f, a1 = 0.0f;
#pragma unroll
  for (int d = 0; d < 16; ++d) {
    float w = shbs[t * 16 + d] * robs[e * 16 + d];
    v2h pv = ((const v2h*)(ckjn + (e * 16 + d) * 64))[lane];
    a0 += w * (float)pv[0];
    a1 += w * (float)pv[1];
  }
  float ss = a0 * a0 + a1 * a1;
#pragma unroll
  for (int m = 1; m < 32; m <<= 1) ss += __shfl_xor(ss, m, 32);
  float s = 1.0f / fmaxf(sqrtf(ss), 1e-12f);
  v2f gx = *(const v2f*)(xk + kn * 64 + 2 * lane);
  atomicAdd(&agg[ej * 64 + 2 * lane],     a0 * s * sigmoidf_(gx[0]));
  atomicAdd(&agg[ej * 64 + 2 * lane + 1], a1 * s * sigmoidf_(gx[1]));
}

// tbw2 = silu(agg) @ W_t3 + b_t3
__global__ void three_lin_kernel(const float* __restrict__ agg,
                                 const _Float16* __restrict__ Wt3p,
                                 const float* __restrict__ b_t3,
                                 float* __restrict__ tbw2, int E) {
  int wave = threadIdx.x >> 5, lane = threadIdx.x & 31;
  long strip = (long)blockIdx.x * (blockDim.x >> 5) + wave;
  long row0 = strip * 16;
  if (row0 >= E) return;
  v8f acc[4];
#pragma unroll
  for (int j = 0; j < 4; ++j) acc[j] = vzero8();
#pragma unroll
  for (int k0 = 0; k0 < 64; k0 += 32) {
    v16h a = load_a_f32<true>(agg, row0, E, 64, k0);
#pragma unroll
    for (int j = 0; j < 4; ++j)
      acc[j] = wmma16(a, load_b_packed(Wt3p, 4, k0, j), acc[j]);
  }
  int n = lane & 15, hi = lane >> 4;
#pragma unroll
  for (int j = 0; j < 4; ++j) {
    int col = j * 16 + n;
    float bias = b_t3[col];
#pragma unroll
    for (int v = 0; v < 8; ++v) {
      long r = row0 + v + 8 * hi;
      if (r < E) tbw2[r * 64 + col] = acc[j][v] + bias;
    }
  }
}

// one wave per edge; lane owns cols 2*lane, 2*lane+1
__global__ void lcao_kernel(const _Float16* __restrict__ cji2,
                            const float* __restrict__ tbw2,
                            const float* __restrict__ robs,
                            float* __restrict__ lcao, int E) {
  int wave = threadIdx.x >> 5, lane = threadIdx.x & 31;
  long e = (long)blockIdx.x * (blockDim.x >> 5) + wave;
  if (e >= E) return;
  v2f wv = *(const v2f*)(tbw2 + e * 64 + 2 * lane);
  float w0 = 1.0f + wv[0], w1 = 1.0f + wv[1];
  float acc0 = 0.0f, acc1 = 0.0f;
#pragma unroll
  for (int d = 0; d < 16; ++d) {
    v2h pv = ((const v2h*)(cji2 + (e * 16 + d) * 64))[lane];
    float v0 = (float)pv[0] * w0;
    float v1 = (float)pv[1] * w1;
    float ss = v0 * v0 + v1 * v1;
#pragma unroll
    for (int m = 1; m < 32; m <<= 1) ss += __shfl_xor(ss, m, 32);
    float s = 1.0f / fmaxf(sqrtf(ss), 1e-12f);
    float r = robs[e * 16 + d];
    acc0 += r * v0 * s;
    acc1 += r * v1 * s;
  }
  float ss2 = acc0 * acc0 + acc1 * acc1;
#pragma unroll
  for (int m = 1; m < 32; m <<= 1) ss2 += __shfl_xor(ss2, m, 32);
  float s2 = 1.0f / fmaxf(sqrtf(ss2), 1e-12f);
  v2f o; o[0] = acc0 * s2; o[1] = acc1 * s2;
  *(v2f*)(lcao + e * 64 + 2 * lane) = o;
}

// m = silu(silu(concat(xq[i],xq[j]))@W_n1+b1)@W_n2+b2 ; scatter lcao*m -> nacc
__global__ void node_msg_kernel(const float* __restrict__ xq,
                                const int* __restrict__ idx_i,
                                const int* __restrict__ idx_j,
                                const _Float16* __restrict__ Wn1p,
                                const float* __restrict__ b_n1,
                                const _Float16* __restrict__ Wn2p,
                                const float* __restrict__ b_n2,
                                const float* __restrict__ lcao,
                                float* __restrict__ nacc, int E) {
  __shared__ alignas(16) _Float16 hbuf[4][16 * 72];
  int wave = threadIdx.x >> 5, lane = threadIdx.x & 31;
  long strip = (long)blockIdx.x * (blockDim.x >> 5) + wave;
  long row0 = strip * 16;
  if (row0 >= E) return;
  _Float16* hb = hbuf[wave];
  int n = lane & 15, hi = lane >> 4;

  v8f acc1[4];
#pragma unroll
  for (int j = 0; j < 4; ++j) acc1[j] = vzero8();
#pragma unroll
  for (int k0 = 0; k0 < 128; k0 += 32) {
    v16h a = load_a_concat_silu(xq, idx_i, idx_j, row0, E, k0);
#pragma unroll
    for (int j = 0; j < 4; ++j)
      acc1[j] = wmma16(a, load_b_packed(Wn1p, 4, k0, j), acc1[j]);
  }
#pragma unroll
  for (int j = 0; j < 4; ++j)
#pragma unroll
    for (int v = 0; v < 8; ++v)
      hb[(v + 8 * hi) * 72 + j * 16 + n] =
          (_Float16)siluf_(acc1[j][v] + b_n1[j * 16 + n]);

  v8f acc2[4];
#pragma unroll
  for (int j = 0; j < 4; ++j) acc2[j] = vzero8();
#pragma unroll
  for (int k0 = 0; k0 < 64; k0 += 32) {
    v16h a = load_a_lds(hb, 72, k0);
#pragma unroll
    for (int j = 0; j < 4; ++j)
      acc2[j] = wmma16(a, load_b_packed(Wn2p, 4, k0, j), acc2[j]);
  }
#pragma unroll
  for (int j = 0; j < 4; ++j)
#pragma unroll
    for (int v = 0; v < 8; ++v) {
      long e = row0 + v + 8 * hi;
      if (e < E) {
        int col = j * 16 + n;
        float mval = acc2[j][v] + b_n2[col];
        float l = lcao[e * 64 + col];
        atomicAdd(&nacc[(long)idx_i[e] * 64 + col], mval * l);
      }
    }
}

// out = x + node_acc @ W_na
__global__ void out_kernel(const float* __restrict__ nacc,
                           const _Float16* __restrict__ Wnap,
                           const float* __restrict__ x, float* __restrict__ out,
                           int N) {
  int wave = threadIdx.x >> 5, lane = threadIdx.x & 31;
  long strip = (long)blockIdx.x * (blockDim.x >> 5) + wave;
  long row0 = strip * 16;
  if (row0 >= N) return;
  v8f acc[8];
#pragma unroll
  for (int j = 0; j < 8; ++j) acc[j] = vzero8();
#pragma unroll
  for (int k0 = 0; k0 < 64; k0 += 32) {
    v16h a = load_a_f32<false>(nacc, row0, N, 64, k0);
#pragma unroll
    for (int j = 0; j < 8; ++j)
      acc[j] = wmma16(a, load_b_packed(Wnap, 8, k0, j), acc[j]);
  }
  int n = lane & 15, hi = lane >> 4;
#pragma unroll
  for (int j = 0; j < 8; ++j)
#pragma unroll
    for (int v = 0; v < 8; ++v) {
      long r = row0 + v + 8 * hi;
      if (r < N) {
        int col = j * 16 + n;
        out[r * 128 + col] = x[r * 128 + col] + acc[j][v];
      }
    }
}

// ================= launch =================

extern "C" void kernel_launch(void* const* d_in, const int* in_sizes, int n_in,
                              void* d_out, int out_size, void* d_ws,
                              size_t ws_size, hipStream_t stream) {
  (void)n_in; (void)out_size; (void)ws_size;
  const float* x    = (const float*)d_in[0];
  const float* cji  = (const float*)d_in[1];
  const float* robs = (const float*)d_in[2];
  const float* shbs = (const float*)d_in[3];
  const int* idx_i  = (const int*)d_in[4];
  const int* idx_j  = (const int*)d_in[5];
  const int* tri_k  = (const int*)d_in[6];
  const int* e_kj   = (const int*)d_in[7];
  const int* e_ji   = (const int*)d_in[8];
  const float* W_nb = (const float*)d_in[9];
  const float* b_nb = (const float*)d_in[10];
  const float* W_c1 = (const float*)d_in[11];
  const float* W_c2 = (const float*)d_in[12];
  const float* W_t3 = (const float*)d_in[13];
  const float* b_t3 = (const float*)d_in[14];
  const float* W_n1 = (const float*)d_in[15];
  const float* b_n1 = (const float*)d_in[16];
  const float* W_n2 = (const float*)d_in[17];
  const float* b_n2 = (const float*)d_in[18];
  const float* W_na = (const float*)d_in[19];
  float* out = (float*)d_out;

  const int N = in_sizes[0] / 128;   // H = 128
  const int E = in_sizes[2] / 16;    // D = 16
  const int T = in_sizes[3] / 16;
  const long rowsED = (long)E * 16;

  char* wsp = (char*)d_ws;
  size_t off = 0;
  auto carve = [&](size_t bytes) -> void* {
    void* p = wsp + off;
    off += (bytes + 255) & ~(size_t)255;
    return p;
  };
  float*    xq   = (float*)carve((size_t)N * 64 * 4);
  float*    xk   = (float*)carve((size_t)N * 64 * 4);
  _Float16* cji2 = (_Float16*)carve((size_t)rowsED * 64 * 2);
  _Float16* ckjn = (_Float16*)carve((size_t)rowsED * 64 * 2);
  float*    agg  = (float*)carve((size_t)E * 64 * 4);
  float*    tbw2 = (float*)carve((size_t)E * 64 * 4);
  float*    lcao = (float*)carve((size_t)E * 64 * 4);
  float*    nacc = (float*)carve((size_t)N * 64 * 4);
  _Float16* Wnbp = (_Float16*)carve(128 * 128 * 2);
  _Float16* Wc1p = (_Float16*)carve(64 * 64 * 2);
  _Float16* Wc2p = (_Float16*)carve(64 * 128 * 2);
  _Float16* Wt3p = (_Float16*)carve(64 * 64 * 2);
  _Float16* Wn1p = (_Float16*)carve(128 * 64 * 2);
  _Float16* Wn2p = (_Float16*)carve(64 * 64 * 2);
  _Float16* Wnap = (_Float16*)carve(64 * 128 * 2);

  // pack weights (waves = ksteps*tiles, 4 waves/block)
  auto packgrid = [](int K, int Nc) { return ((K >> 5) * (Nc >> 4) + 3) / 4; };
  pack_b_kernel<<<packgrid(128, 128), 128, 0, stream>>>(W_nb, Wnbp, 128, 128);
  pack_b_kernel<<<packgrid(64, 64),   128, 0, stream>>>(W_c1, Wc1p, 64, 64);
  pack_b_kernel<<<packgrid(64, 128),  128, 0, stream>>>(W_c2, Wc2p, 64, 128);
  pack_b_kernel<<<packgrid(64, 64),   128, 0, stream>>>(W_t3, Wt3p, 64, 64);
  pack_b_kernel<<<packgrid(128, 64),  128, 0, stream>>>(W_n1, Wn1p, 128, 64);
  pack_b_kernel<<<packgrid(64, 64),   128, 0, stream>>>(W_n2, Wn2p, 64, 64);
  pack_b_kernel<<<packgrid(64, 128),  128, 0, stream>>>(W_na, Wnap, 64, 128);

  {
    int n1 = E * 64, n2 = N * 64;
    zero_kernel<<<(n1 + 255) / 256, 256, 0, stream>>>(agg, n1);
    zero_kernel<<<(n2 + 255) / 256, 256, 0, stream>>>(nacc, n2);
  }
  int stripsN = (N + 15) / 16;
  int stripsE = (E + 15) / 16;
  int stripsC = (int)((rowsED + 15) / 16);

  node_before_kernel<<<(stripsN + 3) / 4, 128, 0, stream>>>(x, Wnbp, b_nb, xq,
                                                            xk, N);
  coeffs_kernel<<<(stripsC + 3) / 4, 128, 0, stream>>>(cji, Wc1p, Wc2p, cji2,
                                                       ckjn, rowsED);
  triplet_kernel<<<(T + 7) / 8, 256, 0, stream>>>(ckjn, robs, shbs, tri_k,
                                                  e_kj, e_ji, xk, agg, T);
  three_lin_kernel<<<(stripsE + 3) / 4, 128, 0, stream>>>(agg, Wt3p, b_t3,
                                                          tbw2, E);
  lcao_kernel<<<(E + 7) / 8, 256, 0, stream>>>(cji2, tbw2, robs, lcao, E);
  node_msg_kernel<<<(stripsE + 3) / 4, 128, 0, stream>>>(
      xq, idx_i, idx_j, Wn1p, b_n1, Wn2p, b_n2, lcao, nacc, E);
  out_kernel<<<(stripsN + 3) / 4, 128, 0, stream>>>(nacc, Wnap, x, out, N);
}
